// graphNet_27066883899548
// MI455X (gfx1250) — compile-verified, matrix operand order checked by reading, not run on hardware
//
#include <hip/hip_runtime.h>
#include <hip/hip_bf16.h>
#include <math.h>

typedef __attribute__((ext_vector_type(16))) __bf16 v16bf;
typedef __attribute__((ext_vector_type(8)))  float  v8f;

#define C_IN 21
#define OUTD 42
#define H1D  672
#define H2D  336
#define H2P  352    // H2 padded to multiple of 32 (GEMM K granularity)
#define NACT 8
#define NGR  8
#define K1P  64     // padded K for layer-1 fused input (21 rel + pad + 21 root + pad)
#define NHP  192    // padded Nout for fused head GEMM (4*42 = 168 -> 192)
#define MAX_LOGSTD 10.0f

// ---------------- degree / inverse ----------------
__global__ __launch_bounds__(256) void k_deg(const int* __restrict__ dst, float* __restrict__ deg, int E) {
  int i = blockIdx.x * 256 + threadIdx.x;
  if (i < E) atomicAdd(&deg[dst[i]], 1.0f);
}

__global__ __launch_bounds__(256) void k_inv(const float* __restrict__ deg, float* __restrict__ inv, int N) {
  int i = blockIdx.x * 256 + threadIdx.x;
  if (i < N) inv[i] = 1.0f / fmaxf(deg[i], 1.0f);
}

// ---------------- weight prep: f32 -> padded, transposed bf16 ----------------
// w1T[672][64]: k<21 -> w1_rel[k][n]; 32<=k<53 -> w1_root[k-32][n]; else 0
__global__ __launch_bounds__(256) void k_prep_w1T(const float* __restrict__ w1_rel,
                                                  const float* __restrict__ w1_root,
                                                  __bf16* __restrict__ w1T) {
  int i = blockIdx.x * 256 + threadIdx.x;
  if (i >= H1D * K1P) return;
  int n = i >> 6, k = i & 63;
  float v = 0.0f;
  if (k < C_IN) v = w1_rel[k * H1D + n];
  else if (k >= 32 && k < 32 + C_IN) v = w1_root[(k - 32) * H1D + n];
  w1T[i] = (__bf16)v;
}

// w2T[672][672]: n<336 -> w2_rel[k][n]; else w2_root[k][n-336]
__global__ __launch_bounds__(256) void k_prep_w2T(const float* __restrict__ w2_rel,
                                                  const float* __restrict__ w2_root,
                                                  __bf16* __restrict__ w2T) {
  int i = blockIdx.x * 256 + threadIdx.x;
  if (i >= H1D * H1D) return;
  int n = i / H1D, k = i - n * H1D;
  float v = (n < H2D) ? w2_rel[k * H2D + n] : w2_root[k * H2D + (n - H2D)];
  w2T[i] = (__bf16)v;
}

// whT[192][352]: k<336 real, k>=336 zero pad; n in 4 blocks of 42 (mu_rel|mu_root|ls_rel|ls_root)
__global__ __launch_bounds__(256) void k_prep_whT(const float* __restrict__ wmu_rel,
                                                  const float* __restrict__ wmu_root,
                                                  const float* __restrict__ wls_rel,
                                                  const float* __restrict__ wls_root,
                                                  __bf16* __restrict__ whT) {
  int i = blockIdx.x * 256 + threadIdx.x;
  if (i >= NHP * H2P) return;
  int n = i / H2P, k = i - n * H2P;
  float v = 0.0f;
  if (k < H2D) {
    if      (n < OUTD)     v = wmu_rel[k * OUTD + n];
    else if (n < 2 * OUTD) v = wmu_root[k * OUTD + (n - OUTD)];
    else if (n < 3 * OUTD) v = wls_rel[k * OUTD + (n - 2 * OUTD)];
    else if (n < 4 * OUTD) v = wls_root[k * OUTD + (n - 3 * OUTD)];
  }
  whT[i] = (__bf16)v;
}

// ---------------- edge scatter-add (segment_sum), strided slices ----------------
__global__ __launch_bounds__(256) void k_scatter_scalar(const float* __restrict__ F, int ldF, int colOff,
                                                        const int* __restrict__ src,
                                                        const int* __restrict__ dst,
                                                        float* __restrict__ out, int ldOut,
                                                        int E, int d) {
  long i = (long)blockIdx.x * 256 + threadIdx.x;
  if (i >= (long)E * d) return;
  int e = (int)(i / d);
  int f = (int)(i - (long)e * d);
  atomicAdd(&out[(long)dst[e] * ldOut + f], F[(long)src[e] * ldF + colOff + f]);
}

__global__ __launch_bounds__(256) void k_scatter_vec4(const float* __restrict__ F, int ldF, int colOff,
                                                      const int* __restrict__ src,
                                                      const int* __restrict__ dst,
                                                      float* __restrict__ out, int ldOut,
                                                      int E, int d4) {
  long i = (long)blockIdx.x * 256 + threadIdx.x;
  if (i >= (long)E * d4) return;
  int e = (int)(i / d4);
  int f = (int)(i - (long)e * d4);
  float4 v = *(const float4*)&F[(long)src[e] * ldF + colOff + f * 4];
  float* o = &out[(long)dst[e] * ldOut + f * 4];
  atomicAdd(o + 0, v.x);
  atomicAdd(o + 1, v.y);
  atomicAdd(o + 2, v.z);
  atomicAdd(o + 3, v.w);
}

// ---------------- layer-1 fused input: in1[N][64] bf16 = [aggx*inv | pad | x | pad] ----------------
__global__ __launch_bounds__(256) void k_in1(const float* __restrict__ aggx, const float* __restrict__ x,
                                             const float* __restrict__ inv, __bf16* __restrict__ in1, int N) {
  long i = (long)blockIdx.x * 256 + threadIdx.x;
  if (i >= (long)N * K1P) return;
  int row = (int)(i >> 6), c = (int)(i & 63);
  float v = 0.0f;
  if (c < C_IN)                      v = aggx[(long)row * C_IN + c] * inv[row];
  else if (c >= 32 && c < 32 + C_IN) v = x[(long)row * C_IN + (c - 32)];
  in1[i] = (__bf16)v;
}

// ---------------- WMMA GEMM: Out[M][Npad] = epi(A[M][K] @ BT[Npad][K]^T + bias) ----------------
// bf16 inputs; K % 32 == 0; Npad % 32 == 0. Double-buffered LDS, one barrier per k-step,
// 2 WMMAs per wave per k-step (wave tile 32x16), global prefetch one step ahead.
// Block = 256 threads (8 waves); block tile 128(M) x 32(N).
__global__ __launch_bounds__(256) void k_gemm_wmma(
    const __bf16* __restrict__ A, int lda,
    const __bf16* __restrict__ BT,          // [Npad][K], row n holds column n of B
    const float* __restrict__ bias, int do_relu,
    void* __restrict__ Out, int out_bf16, int ldo,
    int M, int K, int Npad) {
  __shared__ __align__(16) __bf16 sA[2][128 * 32];  // [row][k]
  __shared__ __align__(16) __bf16 sB[2][32 * 32];   // [n][k]

  const int tid = threadIdx.x;
  const int wave = tid >> 5;
  const int lane = tid & 31;
  const int wr = (wave & 3) * 32;        // wave's 32-row slice
  const int wc = (wave >> 2) * 16;       // wave's 16-col slice
  const int blockRow = blockIdx.x * 128;
  const int blockCol = blockIdx.y * 32;

  // A staging: thread t covers row (t>>1), 16 contiguous bf16 at col (t&1)*16 (two b128)
  const int asrow = tid >> 1;
  const int asoff = (tid & 1) * 16;
  long arow_g = blockRow + asrow;
  if (arow_g >= M) arow_g = M - 1;       // clamp; masked at epilogue
  const __bf16* aptr = A + arow_g * (long)lda + asoff;

  // B staging: thread t covers BT row (blockCol + t>>3), 4 bf16 at (t&7)*4 (one b64)
  const int bsrow = tid >> 3;
  const int bsoff = (tid & 7) * 4;
  const __bf16* bptr = BT + (long)(blockCol + bsrow) * K + bsoff;

  v8f acc0 = {0.f, 0.f, 0.f, 0.f, 0.f, 0.f, 0.f, 0.f};
  v8f acc1 = {0.f, 0.f, 0.f, 0.f, 0.f, 0.f, 0.f, 0.f};
  const int mrow = lane & 15;
  const int kb2 = ((lane < 16) ? 0 : 8) >> 1;  // kbase/2 in u32 units

  // prologue: stage k-step 0 into buffer 0
  uint4 ra0 = *(const uint4*)(aptr);
  uint4 ra1 = *(const uint4*)(aptr + 8);
  uint2 rb  = *(const uint2*)(bptr);
  int buf = 0;
  *(uint4*)&sA[0][asrow * 32 + asoff]     = ra0;
  *(uint4*)&sA[0][asrow * 32 + asoff + 8] = ra1;
  *(uint2*)&sB[0][bsrow * 32 + bsoff]     = rb;
  __syncthreads();

  for (int k0 = 32;; k0 += 32) {
    const bool more = (k0 < K);
    if (more) {
      ra0 = *(const uint4*)(aptr + k0);
      ra1 = *(const uint4*)(aptr + k0 + 8);
      rb  = *(const uint2*)(bptr + k0);
      __builtin_prefetch(aptr + k0 + 32, 0, 0);   // global_prefetch_b8
      __builtin_prefetch(bptr + k0 + 32, 0, 0);
    }

    // fragments from current buffer
    union { v16bf v; unsigned u[8]; } fa0, fa1, fb;
    const unsigned* sAu = (const unsigned*)sA[buf];
    const unsigned* sBu = (const unsigned*)sB[buf];
    const int r0 = (wr + mrow) * 16;
    const int r1 = (wr + 16 + mrow) * 16;
    const int bc = (wc + mrow) * 16;
    #pragma unroll
    for (int j = 0; j < 4; ++j) {
      fa0.u[j]     = sAu[r0 + kb2 + j];
      fa0.u[4 + j] = sAu[r0 + 8 + kb2 + j];
      fa1.u[j]     = sAu[r1 + kb2 + j];
      fa1.u[4 + j] = sAu[r1 + 8 + kb2 + j];
      fb.u[j]      = sBu[bc + kb2 + j];
      fb.u[4 + j]  = sBu[bc + 8 + kb2 + j];
    }
    acc0 = __builtin_amdgcn_wmma_f32_16x16x32_bf16(false, fa0.v, false, fb.v,
                                                   (short)0, acc0, false, false);
    acc1 = __builtin_amdgcn_wmma_f32_16x16x32_bf16(false, fa1.v, false, fb.v,
                                                   (short)0, acc1, false, false);
    if (!more) break;

    // stage next k-step into the other buffer
    buf ^= 1;
    *(uint4*)&sA[buf][asrow * 32 + asoff]     = ra0;
    *(uint4*)&sA[buf][asrow * 32 + asoff + 8] = ra1;
    *(uint2*)&sB[buf][bsrow * 32 + bsoff]     = rb;
    __syncthreads();
  }

  // epilogue: lane holds col n=lane&15; VGPR r -> row r (+8 for upper half-wave)
  const int n = lane & 15;
  const int gcol = blockCol + wc + n;  // always < Npad
  const float bv = bias ? bias[gcol] : 0.0f;
  #pragma unroll
  for (int t = 0; t < 2; ++t) {
    const v8f* accp = t ? &acc1 : &acc0;
    #pragma unroll
    for (int r = 0; r < 8; ++r) {
      int m = r + ((lane < 16) ? 0 : 8);
      int grow = blockRow + wr + t * 16 + m;
      if (grow < M) {
        float v = (*accp)[r] + bv;
        if (do_relu) v = v > 0.0f ? v : 0.0f;
        if (out_bf16) ((__bf16*)Out)[(long)grow * ldo + gcol] = (__bf16)v;
        else          ((float*)Out)[(long)grow * ldo + gcol] = v;
      }
    }
  }
}

// ---------------- h2[N][352] = relu(agg*inv + r2 + b2) -> bf16 (zero pad cols 336..351) ----------------
__global__ __launch_bounds__(256) void k_h2(const float* __restrict__ agg, const float* __restrict__ t2r2,
                                            const float* __restrict__ b2, const float* __restrict__ inv,
                                            __bf16* __restrict__ h2, int N) {
  long i = (long)blockIdx.x * 256 + threadIdx.x;
  if (i >= (long)N * H2P) return;
  int row = (int)(i / H2P);
  int col = (int)(i - (long)row * H2P);
  float v = 0.0f;
  if (col < H2D) {
    long j = (long)row * H2D + col;
    v = agg[j] * inv[row] + t2r2[(long)row * H1D + H2D + col] + b2[col];
    v = v > 0.0f ? v : 0.0f;
  }
  h2[i] = (__bf16)v;
}

// ---------------- z = mu + eps*exp(min(logstd,10)) ----------------
__global__ __launch_bounds__(256) void k_z(const float* __restrict__ aggmu, const float* __restrict__ aggls,
                                           const float* __restrict__ thead,  // [N][192]
                                           const float* __restrict__ bmu, const float* __restrict__ bls,
                                           const float* __restrict__ inv, const float* __restrict__ eps,
                                           float* __restrict__ z, int N) {
  long i = (long)blockIdx.x * 256 + threadIdx.x;
  if (i >= (long)N * OUTD) return;
  int row = (int)(i / OUTD);
  int col = (int)(i - (long)row * OUTD);
  float iv = inv[row];
  const float* th = &thead[(long)row * NHP];
  float mu = aggmu[i] * iv + th[OUTD + col] + bmu[col];
  float ls = fminf(aggls[i] * iv + th[3 * OUTD + col] + bls[col], MAX_LOGSTD);
  z[i] = mu + eps[i] * __expf(ls);
}

// ---------------- p_z = log_softmax(relu(z@wp + bp)) ----------------
__global__ __launch_bounds__(256) void k_pz(const float* __restrict__ z, const float* __restrict__ wp,
                                            const float* __restrict__ bp, float* __restrict__ out, int N) {
  __shared__ float swp[OUTD * C_IN];
  __shared__ float sbp[C_IN];
  for (int i = threadIdx.x; i < OUTD * C_IN; i += 256) swp[i] = wp[i];
  if (threadIdx.x < C_IN) sbp[threadIdx.x] = bp[threadIdx.x];
  __syncthreads();
  int row = blockIdx.x * 256 + threadIdx.x;
  if (row >= N) return;
  float zr[OUTD];
  #pragma unroll
  for (int j = 0; j < OUTD; ++j) zr[j] = z[(long)row * OUTD + j];
  float v[C_IN];
  #pragma unroll
  for (int c = 0; c < C_IN; ++c) {
    float s = sbp[c];
    for (int j = 0; j < OUTD; ++j) s += zr[j] * swp[j * C_IN + c];
    v[c] = s > 0.0f ? s : 0.0f;
  }
  float m = v[0];
  #pragma unroll
  for (int c = 1; c < C_IN; ++c) m = fmaxf(m, v[c]);
  float lse = 0.0f;
  #pragma unroll
  for (int c = 0; c < C_IN; ++c) lse += __expf(v[c] - m);
  lse = __logf(lse);
  #pragma unroll
  for (int c = 0; c < C_IN; ++c) out[(long)row * C_IN + c] = v[c] - m - lse;
}

// ---------------- graph pooling (two-stage) ----------------
__global__ __launch_bounds__(256) void k_pool(const float* __restrict__ z, const int* __restrict__ batch,
                                              float* __restrict__ pooled, float* __restrict__ cnt, int N) {
  __shared__ float sp[NGR * OUTD];
  __shared__ float sc[NGR];
  for (int i = threadIdx.x; i < NGR * OUTD; i += 256) sp[i] = 0.0f;
  if (threadIdx.x < NGR) sc[threadIdx.x] = 0.0f;
  __syncthreads();
  int i = blockIdx.x * 256 + threadIdx.x;
  if (i < N) {
    int b = batch[i];
    atomicAdd(&sc[b], 1.0f);
    for (int j = 0; j < OUTD; ++j) atomicAdd(&sp[b * OUTD + j], z[(long)i * OUTD + j]);
  }
  __syncthreads();
  for (int t = threadIdx.x; t < NGR * OUTD; t += 256) atomicAdd(&pooled[t], sp[t]);
  if (threadIdx.x < NGR) atomicAdd(&cnt[threadIdx.x], sc[threadIdx.x]);
}

__global__ void k_az(const float* __restrict__ pooled, const float* __restrict__ cnt,
                     const float* __restrict__ wa, const float* __restrict__ ba,
                     float* __restrict__ out) {
  int g = threadIdx.x;
  if (g >= NGR) return;
  float c = fmaxf(cnt[g], 1.0f);
  float v[NACT];
  #pragma unroll
  for (int a = 0; a < NACT; ++a) {
    float s = ba[a];
    for (int j = 0; j < OUTD; ++j) s += (pooled[g * OUTD + j] / c) * wa[j * NACT + a];
    v[a] = s;
  }
  float m = v[0];
  #pragma unroll
  for (int a = 1; a < NACT; ++a) m = fmaxf(m, v[a]);
  float lse = 0.0f;
  #pragma unroll
  for (int a = 0; a < NACT; ++a) lse += __expf(v[a] - m);
  lse = __logf(lse);
  #pragma unroll
  for (int a = 0; a < NACT; ++a) out[g * NACT + a] = v[a] - m - lse;
}

// ---------------- host-side orchestration ----------------
static inline int cdiv(long a, long b) { return (int)((a + b - 1) / b); }

extern "C" void kernel_launch(void* const* d_in, const int* in_sizes, int n_in,
                              void* d_out, int out_size, void* d_ws, size_t ws_size,
                              hipStream_t stream) {
  const float* x        = (const float*)d_in[0];
  const int*   eidx     = (const int*)d_in[1];
  const int*   batch    = (const int*)d_in[2];
  const float* eps      = (const float*)d_in[3];
  const float* w1_rel   = (const float*)d_in[4];
  const float* w1_root  = (const float*)d_in[5];
  const float* b1       = (const float*)d_in[6];
  const float* w2_rel   = (const float*)d_in[7];
  const float* w2_root  = (const float*)d_in[8];
  const float* b2       = (const float*)d_in[9];
  const float* wmu_rel  = (const float*)d_in[10];
  const float* wmu_root = (const float*)d_in[11];
  const float* bmu      = (const float*)d_in[12];
  const float* wls_rel  = (const float*)d_in[13];
  const float* wls_root = (const float*)d_in[14];
  const float* bls      = (const float*)d_in[15];
  const float* wp       = (const float*)d_in[16];
  const float* bp       = (const float*)d_in[17];
  const float* wa       = (const float*)d_in[18];
  const float* ba       = (const float*)d_in[19];

  const int N = in_sizes[2];
  const int E = in_sizes[1] / 2;
  const int* src = eidx;
  const int* dst = eidx + E;

  // workspace carve-up (256B aligned)
  char* ws = (char*)d_ws;
  size_t off = 0;
  auto carve = [&](size_t bytes) -> char* {
    char* p = ws + off;
    off = (off + bytes + 255) & ~(size_t)255;
    return p;
  };
  float*  deg    = (float*)carve((size_t)N * 4);
  float*  inv    = (float*)carve((size_t)N * 4);
  float*  aggx   = (float*)carve((size_t)N * C_IN * 4);
  __bf16* in1    = (__bf16*)carve(((size_t)N + 128) * K1P * 2);   // +pad rows for clamp safety
  __bf16* h1     = (__bf16*)carve(((size_t)N + 128) * H1D * 2);
  float*  t2r2   = (float*)carve((size_t)N * H1D * 4);            // [t2 | r2]
  float*  aggb   = (float*)carve((size_t)N * H2D * 4);            // aggt2, later aggmu/aggls
  __bf16* h2     = (__bf16*)carve(((size_t)N + 128) * H2P * 2);
  float*  thead  = (float*)carve((size_t)N * NHP * 4);            // [tmu|rmu|tls|rls|pad]
  __bf16* w1T    = (__bf16*)carve((size_t)H1D * K1P * 2);
  __bf16* w2T    = (__bf16*)carve((size_t)H1D * H1D * 2);
  __bf16* whT    = (__bf16*)carve((size_t)NHP * H2P * 2);
  float*  pooled = (float*)carve((size_t)(NGR * OUTD + NGR) * 4);
  float*  cnt    = pooled + NGR * OUTD;

  float* aggmu = aggb;
  float* aggls = aggb + (size_t)N * OUTD;

  // output layout: p_z [N*21], a_z [64], z [N*42]
  float* out_pz = (float*)d_out;
  float* out_az = out_pz + (size_t)N * C_IN;
  float* out_z  = out_az + NGR * NACT;

  // --- zero accumulators (graph-replay safe) ---
  hipMemsetAsync(deg, 0, (size_t)N * 4, stream);
  hipMemsetAsync(aggx, 0, (size_t)N * C_IN * 4, stream);
  hipMemsetAsync(pooled, 0, (size_t)(NGR * OUTD + NGR) * 4, stream);

  // --- weight prep (bf16, transposed, padded) ---
  k_prep_w1T<<<cdiv(H1D * K1P, 256), 256, 0, stream>>>(w1_rel, w1_root, w1T);
  k_prep_w2T<<<cdiv(H1D * H1D, 256), 256, 0, stream>>>(w2_rel, w2_root, w2T);
  k_prep_whT<<<cdiv(NHP * H2P, 256), 256, 0, stream>>>(wmu_rel, wmu_root, wls_rel, wls_root, whT);

  // --- degree / inverse ---
  k_deg<<<cdiv(E, 256), 256, 0, stream>>>(dst, deg, E);
  k_inv<<<cdiv(N, 256), 256, 0, stream>>>(deg, inv, N);

  // --- layer 1: agg x (21 dims), build fused bf16 input, one GEMM -> h1 ---
  k_scatter_scalar<<<cdiv((long)E * C_IN, 256), 256, 0, stream>>>(x, C_IN, 0, src, dst, aggx, C_IN, E, C_IN);
  k_in1<<<cdiv((long)N * K1P, 256), 256, 0, stream>>>(aggx, x, inv, in1, N);
  {
    dim3 g(cdiv(N, 128), H1D / 32);
    k_gemm_wmma<<<g, 256, 0, stream>>>(in1, K1P, w1T, b1, 1, h1, 1, H1D, N, K1P, H1D);
  }

  // --- layer 2: fused rel|root GEMM, then aggregate 336 dims ---
  {
    dim3 g(cdiv(N, 128), H1D / 32);
    k_gemm_wmma<<<g, 256, 0, stream>>>(h1, H1D, w2T, nullptr, 0, t2r2, 0, H1D, N, H1D, H1D);
  }
  hipMemsetAsync(aggb, 0, (size_t)N * H2D * 4, stream);
  k_scatter_vec4<<<cdiv((long)E * (H2D / 4), 256), 256, 0, stream>>>(
      t2r2, H1D, 0, src, dst, aggb, H2D, E, H2D / 4);
  k_h2<<<cdiv((long)N * H2P, 256), 256, 0, stream>>>(aggb, t2r2, b2, inv, h2, N);

  // --- heads: one fused GEMM [N,352]->[N,192], aggregate 2x42 dims ---
  {
    dim3 g(cdiv(N, 128), NHP / 32);
    k_gemm_wmma<<<g, 256, 0, stream>>>(h2, H2P, whT, nullptr, 0, thead, 0, NHP, N, H2P, NHP);
  }
  hipMemsetAsync(aggb, 0, 2 * (size_t)N * OUTD * 4, stream);
  k_scatter_scalar<<<cdiv((long)E * OUTD, 256), 256, 0, stream>>>(thead, NHP, 0, src, dst, aggmu, OUTD, E, OUTD);
  k_scatter_scalar<<<cdiv((long)E * OUTD, 256), 256, 0, stream>>>(thead, NHP, 2 * OUTD, src, dst, aggls, OUTD, E, OUTD);

  // --- z (straight into d_out), then heads over z ---
  k_z<<<cdiv((long)N * OUTD, 256), 256, 0, stream>>>(aggmu, aggls, thead, bmu, bls, inv, eps, out_z, N);
  k_pz<<<cdiv(N, 256), 256, 0, stream>>>(out_z, wp, bp, out_pz, N);
  k_pool<<<cdiv(N, 256), 256, 0, stream>>>(out_z, batch, pooled, cnt, N);
  k_az<<<1, 32, 0, stream>>>(pooled, cnt, wa, ba, out_az);
}
